// GRUNet_42236708389192
// MI455X (gfx1250) — compile-verified, matrix operand order checked by reading, not run on hardware
//
#include <hip/hip_runtime.h>

// MI455X / gfx1250 fused 2-layer GRU (H=32, IN=3, T=512, B=4096) + FC + softmax.
//
// ~40 GFLOP total, ~25 MB mandatory HBM traffic -> memory time ~1us at 23.3 TB/s;
// runtime is the serial T=512 recurrence. Design: recurrence fully register-
// resident per wave (16 batch rows), WMMA for every matmul, VALU activations
// co-executing with XDL WMMA.
//
// Transposed formulation: gates^T[g][b] = W[g][k] x h^T[k][b]. Weights are
// loop-invariant WMMA A-fragments in VGPRs; hidden state h^T is the B-fragment.
// CDNA5 layout property: f32 C/D layout (reg r <-> row r+8*half, lane <-> col)
// packed as {cvt_f16(Cfrag0), cvt_f16(Cfrag1)} is EXACTLY a legal f16
// B-fragment => C->B transpose needs zero cross-lane moves, no LDS.
//
// Software pipelining: layer1(t) and layer0(t+1) both depend only on h1(t), so
// each loop iteration issues both layers' WMMAs (12 + 6x16x16x4 + 6) before
// either activation block, halving the per-step critical path.
//
// r/z gate weights are pre-scaled by 0.5 so sigmoid(x)=0.5*tanh(0.5x)+0.5
// needs only v_tanh_f32 + v_fma_f32 (hardware tanh confirmed on gfx1250).

typedef __attribute__((ext_vector_type(16))) _Float16 v16h;
typedef __attribute__((ext_vector_type(8)))  float    v8f;
typedef __attribute__((ext_vector_type(2)))  float    v2f;

#define WMMA16(A, B, C) __builtin_amdgcn_wmma_f32_16x16x32_f16(false, (A), false, (B), (short)0, (C), false, false)
#define WMMA4(A, B, C)  __builtin_amdgcn_wmma_f32_16x16x4_f32 (false, (A), false, (B), (short)0, (C), false, false)

#define T_STEPS 512
#define HID 32

#if __has_builtin(__builtin_amdgcn_tanhf)
__device__ __forceinline__ float tanh_hw(float v) { return __builtin_amdgcn_tanhf(v); }
#else
__device__ __forceinline__ float tanh_hw(float v) {
    float e = __expf(2.0f * v);
    return (e - 1.0f) * __builtin_amdgcn_rcpf(e + 1.0f);
}
#endif
// Input already pre-scaled by 0.5 via the weight fragments.
__device__ __forceinline__ float sig_pre(float a) {
    return __builtin_fmaf(0.5f, tanh_hw(a), 0.5f);
}

__global__ __launch_bounds__(128)
void gru2_fused_wmma(const float* __restrict__ x,
                     const float* __restrict__ w_ih0, const float* __restrict__ w_hh0,
                     const float* __restrict__ b_ih0, const float* __restrict__ b_hh0,
                     const float* __restrict__ w_ih1, const float* __restrict__ w_hh1,
                     const float* __restrict__ b_ih1, const float* __restrict__ b_hh1,
                     const float* __restrict__ fc_w,  const float* __restrict__ fc_b,
                     float* __restrict__ out)
{
    const int lane = threadIdx.x & 31;
    const int wv   = threadIdx.x >> 5;
    const int nl   = lane & 15;      // batch column within tile / A row index
    const int hi   = lane >> 4;      // wave half selects K/M sub-range per ISA layout
    const int base = (blockIdx.x * 4 + wv) * 16;

    // ------------- loop-invariant weight fragments (A-matrices) -------------
    // Gate fragment order: r(0,1) z(2,3) n(4,5). r/z fragments pre-scaled 0.5.
    v16h whh0[6], wih1[6], whh1[6];
#pragma unroll
    for (int f = 0; f < 6; ++f) {
        const int g = f * 16 + nl;
        const float s = (f < 4) ? 0.5f : 1.0f;
        v16h a, b, c;
#pragma unroll
        for (int e = 0; e < 16; ++e) {
            const int k = e + ((e < 8) ? 0 : 8) + hi * 8;
            a[e] = (_Float16)(w_hh0[g * HID + k] * s);
            b[e] = (_Float16)(w_ih1[g * HID + k] * s);
            c[e] = (_Float16)(w_hh1[g * HID + k] * s);
        }
        whh0[f] = a; wih1[f] = b; whh1[f] = c;
    }
    // f32 16x4 A-fragments for w_ih0 (K=3 padded to 4; col 3 carries the layer-0
    // biases, paired with the constant-1 row of the x B-fragment).
    v2f wih0[6];
#pragma unroll
    for (int f = 0; f < 6; ++f) {
        const int g = f * 16 + nl;
        const float s = (f < 4) ? 0.5f : 1.0f;
        const float c3 = (f < 4) ? (b_ih0[g] + b_hh0[g]) : b_ih0[g];
        v2f w;
        w.x = (hi ? w_ih0[g * 3 + 2] : w_ih0[g * 3 + 0]) * s;
        w.y = (hi ? c3               : w_ih0[g * 3 + 1]) * s;
        wih0[f] = w;
    }
    // Bias C-init fragments (C layout: reg r -> row gbase + r + hi*8)
    v8f bhh0n[2], bsum1[4], bih1n[2], bhh1n[2];
#pragma unroll
    for (int f = 0; f < 2; ++f)
#pragma unroll
        for (int r = 0; r < 8; ++r) {
            const int g = 64 + f * 16 + r + hi * 8;
            bhh0n[f][r] = b_hh0[g];
            bih1n[f][r] = b_ih1[g];
            bhh1n[f][r] = b_hh1[g];
        }
#pragma unroll
    for (int f = 0; f < 4; ++f)
#pragma unroll
        for (int r = 0; r < 8; ++r) {
            const int g = f * 16 + r + hi * 8;
            bsum1[f][r] = 0.5f * (b_ih1[g] + b_hh1[g]);
        }

    // Hidden state: f32 C-layout + same values packed as f16 B-fragments.
    v8f  h1c0 = {}, h1c1 = {}, h2c0 = {}, h2c1 = {};
    v16h h1b = {}, h2b = {};

    // Per-lane x pointers, advanced by one timestep (3 floats) per iteration.
    const float* xpA = x + (size_t)(base + nl) * (T_STEPS * 3) + 2 * hi;
    const float* xpB = x + (size_t)(base + nl) * (T_STEPS * 3) + 1;

    // Layer accumulators (two independent sets so both layers stay in flight).
    v8f A0, A1, A2, A3, Ani0, Ani1, Anh0, Anh1;   // layer 0
    v8f B0, B1, B2, B3, Bni0, Bni1, Bnh0, Bnh1;   // layer 1

    auto l0_mm = [&](float xa, float xb) {
        v2f xB; xB.x = xa; xB.y = hi ? 1.0f : xb;
        v8f z8 = {};
        A0   = WMMA4(wih0[0], xB, z8);
        A1   = WMMA4(wih0[1], xB, z8);
        A2   = WMMA4(wih0[2], xB, z8);
        A3   = WMMA4(wih0[3], xB, z8);
        Ani0 = WMMA4(wih0[4], xB, z8);
        Ani1 = WMMA4(wih0[5], xB, z8);
        A0   = WMMA16(whh0[0], h1b, A0);
        A1   = WMMA16(whh0[1], h1b, A1);
        A2   = WMMA16(whh0[2], h1b, A2);
        A3   = WMMA16(whh0[3], h1b, A3);
        Anh0 = WMMA16(whh0[4], h1b, bhh0n[0]);
        Anh1 = WMMA16(whh0[5], h1b, bhh0n[1]);
    };
    auto l0_act = [&]() {
#pragma unroll
        for (int r = 0; r < 8; ++r) {
            float rv = sig_pre(A0[r]);
            float zv = sig_pre(A2[r]);
            float nv = tanh_hw(__builtin_fmaf(rv, Anh0[r], Ani0[r]));
            h1c0[r] = __builtin_fmaf(zv, h1c0[r] - nv, nv);
            rv = sig_pre(A1[r]);
            zv = sig_pre(A3[r]);
            nv = tanh_hw(__builtin_fmaf(rv, Anh1[r], Ani1[r]));
            h1c1[r] = __builtin_fmaf(zv, h1c1[r] - nv, nv);
        }
#pragma unroll
        for (int r = 0; r < 8; ++r) { h1b[r] = (_Float16)h1c0[r]; h1b[8 + r] = (_Float16)h1c1[r]; }
    };
    auto l1_mm = [&]() {
        B0   = WMMA16(wih1[0], h1b, bsum1[0]);
        B1   = WMMA16(wih1[1], h1b, bsum1[1]);
        B2   = WMMA16(wih1[2], h1b, bsum1[2]);
        B3   = WMMA16(wih1[3], h1b, bsum1[3]);
        Bni0 = WMMA16(wih1[4], h1b, bih1n[0]);
        Bni1 = WMMA16(wih1[5], h1b, bih1n[1]);
        B0   = WMMA16(whh1[0], h2b, B0);
        B1   = WMMA16(whh1[1], h2b, B1);
        B2   = WMMA16(whh1[2], h2b, B2);
        B3   = WMMA16(whh1[3], h2b, B3);
        Bnh0 = WMMA16(whh1[4], h2b, bhh1n[0]);
        Bnh1 = WMMA16(whh1[5], h2b, bhh1n[1]);
    };
    auto l1_act = [&]() {
#pragma unroll
        for (int r = 0; r < 8; ++r) {
            float rv = sig_pre(B0[r]);
            float zv = sig_pre(B2[r]);
            float nv = tanh_hw(__builtin_fmaf(rv, Bnh0[r], Bni0[r]));
            h2c0[r] = __builtin_fmaf(zv, h2c0[r] - nv, nv);
            rv = sig_pre(B1[r]);
            zv = sig_pre(B3[r]);
            nv = tanh_hw(__builtin_fmaf(rv, Bnh1[r], Bni1[r]));
            h2c1[r] = __builtin_fmaf(zv, h2c1[r] - nv, nv);
        }
#pragma unroll
        for (int r = 0; r < 8; ++r) { h2b[r] = (_Float16)h2c0[r]; h2b[8 + r] = (_Float16)h2c1[r]; }
    };

    // Prologue: h1(0)
    l0_mm(xpA[0], xpB[0]);
    l0_act();

    // Pipelined main loop: layer1(t) and layer0(t+1) are independent (both need
    // only h1(t)) -> issue all WMMAs of both before either activation block.
#pragma unroll 1
    for (int t = 0; t < T_STEPS - 1; ++t) {
        xpA += 3; xpB += 3;
        const float xa = *xpA;
        const float xb = *xpB;
        l1_mm();            // reads h1b(t), h2b(t)
        l0_mm(xa, xb);      // reads h1b(t), x(t+1)
        l1_act();           // h2(t) -> h2(t+1)... (writes h2c/h2b)
        l0_act();           // h1(t) -> h1(t+1)   (writes h1c/h1b)
    }
    // Epilogue: final layer1 step
    l1_mm();
    l1_act();

    // ---------------- FC head + softmax (tiny; one LDS transpose) ----------------
    __shared__ float hbuf[4][16][33];
#pragma unroll
    for (int r = 0; r < 8; ++r) {
        hbuf[wv][nl][r + hi * 8]      = h2c0[r];
        hbuf[wv][nl][16 + r + hi * 8] = h2c1[r];
    }
    __syncthreads();
    if (lane < 16) {
        const int b = base + lane;
        float l0 = fc_b[0], l1 = fc_b[1], l2 = fc_b[2];
#pragma unroll
        for (int j = 0; j < HID; ++j) {
            const float hv = hbuf[wv][lane][j];
            l0 = __builtin_fmaf(fc_w[j], hv, l0);
            l1 = __builtin_fmaf(fc_w[HID + j], hv, l1);
            l2 = __builtin_fmaf(fc_w[2 * HID + j], hv, l2);
        }
        const float m  = fmaxf(l0, fmaxf(l1, l2));
        const float e0 = __expf(l0 - m), e1 = __expf(l1 - m), e2 = __expf(l2 - m);
        const float inv = __builtin_amdgcn_rcpf(e0 + e1 + e2);
        out[(size_t)b * 3 + 0] = e0 * inv;
        out[(size_t)b * 3 + 1] = e1 * inv;
        out[(size_t)b * 3 + 2] = e2 * inv;
    }
}

extern "C" void kernel_launch(void* const* d_in, const int* in_sizes, int n_in,
                              void* d_out, int out_size, void* d_ws, size_t ws_size,
                              hipStream_t stream) {
    const float* x     = (const float*)d_in[0];
    const float* w_ih0 = (const float*)d_in[1];
    const float* w_hh0 = (const float*)d_in[2];
    const float* b_ih0 = (const float*)d_in[3];
    const float* b_hh0 = (const float*)d_in[4];
    const float* w_ih1 = (const float*)d_in[5];
    const float* w_hh1 = (const float*)d_in[6];
    const float* b_ih1 = (const float*)d_in[7];
    const float* b_hh1 = (const float*)d_in[8];
    const float* fc_w  = (const float*)d_in[9];
    const float* fc_b  = (const float*)d_in[10];
    float* out = (float*)d_out;

    const int B = in_sizes[0] / (T_STEPS * 3);   // 4096
    const int blocks = B / 64;                   // 4 waves x 16 batch rows per block
    hipLaunchKernelGGL(gru2_fused_wmma, dim3(blocks), dim3(128), 0, stream,
                       x, w_ih0, w_hh0, b_ih0, b_hh0,
                       w_ih1, w_hh1, b_ih1, b_hh1, fc_w, fc_b, out);
}
